// MyNewGCN_25890062860857
// MI455X (gfx1250) — compile-verified, matrix operand order checked by reading, not run on hardware
//
#include <hip/hip_runtime.h>

typedef __attribute__((ext_vector_type(16))) _Float16 v16h;
typedef __attribute__((ext_vector_type(8)))  float    v8f;

#define N_SU_  8304
#define N_ACE_ 35760
#define N_NMF_ 48600
#define N_WAT_ 59136
#define MAXN_  59136

// ---------------- utility: zero a float buffer ----------------
__global__ void zero_f32(float* __restrict__ p, int n) {
  int i = blockIdx.x * blockDim.x + threadIdx.x;
  if (i < n) p[i] = 0.0f;
}

// ---------------- dense GEMM via WMMA: C(n x NCOL) = X(n x K) @ W(K x NCOL) ----
// K multiple of 32, NCOL multiple of 16. W staged in LDS PRE-PACKED in the
// WMMA B-fragment layout: fragment (kt, t) lane l owns 16 contiguous halfs at
// ((kt*NT + t)*32 + l)*16  -> two ds_load_b128 per fragment, no repacking.
// One wave per 16-row tile, 8 waves (256 thr) per workgroup -> 128 rows/WG.
template<int K, int NCOL>
__global__ void gemm_wmma(const float* __restrict__ X, const float* __restrict__ W,
                          float* __restrict__ C, int n) {
  constexpr int KT = K / 32;
  constexpr int NT = NCOL / 16;
  __shared__ __align__(32) _Float16 w_lds[K * NCOL];

  for (int i = threadIdx.x; i < K * NCOL; i += blockDim.x) {
    const int k = i / NCOL, c = i % NCOL;
    const int kt = k >> 5, krem = k & 31;
    const int t = c >> 4,  m = c & 15;
    const int hi = (krem >> 3) & 1;                    // half-wave selector
    const int j  = (krem & 7) + ((krem >> 4) << 3);    // slot within 16 halfs
    w_lds[(((kt * NT + t) * 32) + (hi * 16 + m)) * 16 + j] = (_Float16)W[i];
  }
  __syncthreads();

  const int wave = threadIdx.x >> 5;
  const int lane = threadIdx.x & 31;
  const int r0 = (blockIdx.x * 8 + wave) * 16;
  if (r0 >= n) return;                       // wave-uniform exit

  const int m  = lane & 15;
  const int hi = lane >> 4;
  int row = r0 + m; if (row >= n) row = n - 1;   // clamp tail loads

  v8f acc[NT] = {};

  for (int kt = 0; kt < KT; ++kt) {
    const int kb = kt * 32 + hi * 8;         // 32B-aligned K base per half-wave
    const float4* xp = (const float4*)(X + (long long)row * K + kb);
    const float4 x0 = xp[0], x1 = xp[1];     // K = kb .. kb+7
    const float4 x2 = xp[4], x3 = xp[5];     // K = kb+16 .. kb+23
    v16h a;
    a[0]  = (_Float16)x0.x; a[1]  = (_Float16)x0.y;
    a[2]  = (_Float16)x0.z; a[3]  = (_Float16)x0.w;
    a[4]  = (_Float16)x1.x; a[5]  = (_Float16)x1.y;
    a[6]  = (_Float16)x1.z; a[7]  = (_Float16)x1.w;
    a[8]  = (_Float16)x2.x; a[9]  = (_Float16)x2.y;
    a[10] = (_Float16)x2.z; a[11] = (_Float16)x2.w;
    a[12] = (_Float16)x3.x; a[13] = (_Float16)x3.y;
    a[14] = (_Float16)x3.z; a[15] = (_Float16)x3.w;

#pragma unroll
    for (int t = 0; t < NT; ++t) {
      const v16h b =
          *reinterpret_cast<const v16h*>(&w_lds[((kt * NT + t) * 32 + lane) * 16]);
      acc[t] = __builtin_amdgcn_wmma_f32_16x16x32_f16(
          false, a, false, b, (short)0, acc[t], false, false);
    }
  }

  // D layout: lane%16 = column; VGPR i -> M = i + 8*(lane>=16)
#pragma unroll
  for (int t = 0; t < NT; ++t) {
    const int c = t * 16 + m;
#pragma unroll
    for (int i = 0; i < 8; ++i) {
      const int rr = r0 + hi * 8 + i;
      if (rr < n) C[(long long)rr * NCOL + c] = acc[t][i];
    }
  }
}

// ---------------- COO scatter-add: dst[row[e],f] += val[e]*src[col[e],f] ------
__global__ void scatter_add(const int* __restrict__ row, const int* __restrict__ col,
                            const float* __restrict__ val, const float* __restrict__ src,
                            float* __restrict__ dst, long long total, int logF) {
  long long t = (long long)blockIdx.x * blockDim.x + threadIdx.x;
  if (t >= total) return;
  const int F = 1 << logF;
  const long long e = t >> logF;
  const int f = (int)(t & (F - 1));
  const float x = val[e] * src[(long long)col[e] * F + f];
  atomicAdd(&dst[(long long)row[e] * F + f], x);
}

// ---------------- bias + relu, in place, F = 64 -------------------------------
__global__ void bias_relu64(float* __restrict__ h, const float* __restrict__ b, int n64) {
  int i = blockIdx.x * blockDim.x + threadIdx.x;
  if (i < n64) {
    float x = h[i] + b[i & 63];
    h[i] = x > 0.0f ? x : 0.0f;
  }
}

// ---------------- block pooling: out[b][f] = sum_r z[b*rows+r][f] + rows*b2[f] -
__global__ void pool_kernel(const float* __restrict__ z, const float* __restrict__ b2,
                            float* __restrict__ out, int blockRows) {
  __shared__ float red[256];
  const int b = blockIdx.x;
  const int f = threadIdx.x & 15, s = threadIdx.x >> 4;
  float acc = 0.0f;
  for (int r = s; r < blockRows; r += 16)
    acc += z[((long long)b * blockRows + r) * 16 + f];
  red[threadIdx.x] = acc;
  __syncthreads();
  if (s == 0) {
    float sum = 0.0f;
#pragma unroll
    for (int k = 0; k < 16; ++k) sum += red[k * 16 + f];
    out[b * 16 + f] = sum + (float)blockRows * b2[f];
  }
}

// ---------------- 3 heads x 4 rows, MLP 101->32->16->8->1 ---------------------
__global__ void heads_kernel(const float* __restrict__ pools,
                             const float* __restrict__ solute_env,
                             const float* __restrict__ ace_env,
                             const float* __restrict__ nmf_env,
                             const float* __restrict__ wat_env,
                             const float* __restrict__ fc1w, const float* __restrict__ fc1b,
                             const float* __restrict__ fc2w, const float* __restrict__ fc2b,
                             const float* __restrict__ fc3w, const float* __restrict__ fc3b,
                             const float* __restrict__ fc4w, const float* __restrict__ fc4b,
                             float* __restrict__ out) {
  const int r = threadIdx.x;
  if (r >= 12) return;
  const int h = r >> 2, q = r & 3;
  const float extras[3][5] = {
      {-1.3711368f, -0.813852f,  -7.31e-05f, 65.1321f,  84.32505f},
      {-1.6522485f, -0.2748812f, -7.81e-05f, 83.89315f, 101.94061f},
      { 0.4208888f,  0.5955439f,  0.0f,      28.78463f, 45.70048f}};
  const float* su  = pools + h * 64;         // slots 0..2: su_ace, su_nmf, su_wat
  const float* sv  = pools + 192 + h * 64;   // slots 3..5: sv_ace, sv_nmf, sv_wat
  const float* env = (h == 0) ? ace_env : ((h == 1) ? nmf_env : wat_env);
  float x[101];
  for (int i = 0; i < 16; ++i) x[i]      = su[q * 16 + i];
  for (int i = 0; i < 16; ++i) x[16 + i] = sv[q * 16 + i];
  for (int i = 0; i < 32; ++i) x[32 + i] = solute_env[i];
  for (int i = 0; i < 32; ++i) x[64 + i] = env[i];
  for (int i = 0; i < 5;  ++i) x[96 + i] = extras[h][i];
  float a1[32];
  for (int j = 0; j < 32; ++j) {
    float s = fc1b[j];
    for (int k = 0; k < 101; ++k) s += x[k] * fc1w[k * 32 + j];
    a1[j] = s > 0.0f ? s : 0.0f;
  }
  float a2[16];
  for (int j = 0; j < 16; ++j) {
    float s = fc2b[j];
    for (int k = 0; k < 32; ++k) s += a1[k] * fc2w[k * 16 + j];
    a2[j] = s > 0.0f ? s : 0.0f;
  }
  float a3[8];
  for (int j = 0; j < 8; ++j) {
    float s = fc3b[j];
    for (int k = 0; k < 16; ++k) s += a2[k] * fc3w[k * 8 + j];
    a3[j] = s > 0.0f ? s : 0.0f;
  }
  float s = fc4b[0];
  for (int k = 0; k < 8; ++k) s += a3[k] * fc4w[k];
  out[r] = s;
}

// ------------------------------------------------------------------------------
extern "C" void kernel_launch(void* const* d_in, const int* in_sizes, int n_in,
                              void* d_out, int out_size, void* d_ws, size_t ws_size,
                              hipStream_t stream) {
  (void)in_sizes; (void)n_in; (void)out_size; (void)ws_size;

  const float* X_su_ace = (const float*)d_in[0];
  const float* X_sv_ace = (const float*)d_in[1];
  const float* X_su_nmf = (const float*)d_in[2];
  const float* X_sv_nmf = (const float*)d_in[3];
  const float* X_su_wat = (const float*)d_in[4];
  const float* X_sv_wat = (const float*)d_in[5];
  const int *su_row = (const int*)d_in[6],  *su_col = (const int*)d_in[7];
  const float* su_val = (const float*)d_in[8];
  const int *ace_row = (const int*)d_in[9], *ace_col = (const int*)d_in[10];
  const float* ace_val = (const float*)d_in[11];
  const int *nmf_row = (const int*)d_in[12], *nmf_col = (const int*)d_in[13];
  const float* nmf_val = (const float*)d_in[14];
  const int *wat_row = (const int*)d_in[15], *wat_col = (const int*)d_in[16];
  const float* wat_val = (const float*)d_in[17];
  const float* solute_env = (const float*)d_in[18];
  const float* ace_env    = (const float*)d_in[19];
  const float* nmf_env    = (const float*)d_in[20];
  const float* wat_env    = (const float*)d_in[21];
  const float* W1 = (const float*)d_in[22]; const float* b1 = (const float*)d_in[23];
  const float* W2 = (const float*)d_in[24]; const float* b2 = (const float*)d_in[25];
  const float* fc1w = (const float*)d_in[26]; const float* fc1b = (const float*)d_in[27];
  const float* fc2w = (const float*)d_in[28]; const float* fc2b = (const float*)d_in[29];
  const float* fc3w = (const float*)d_in[30]; const float* fc3b = (const float*)d_in[31];
  const float* fc4w = (const float*)d_in[32]; const float* fc4b = (const float*)d_in[33];

  float* ws    = (float*)d_ws;
  float* xw1   = ws;                                       // MAXN*64
  float* s1h   = ws + (size_t)MAXN_ * 64;                  // MAXN*64
  float* hw2   = ws + (size_t)MAXN_ * 128;                 // MAXN*16
  float* s2    = ws + (size_t)MAXN_ * 128 + (size_t)MAXN_ * 16;  // MAXN*16
  float* pools = ws + (size_t)MAXN_ * 128 + (size_t)MAXN_ * 32;  // 6*64

  auto run = [&](const float* X, const int* er, const int* ec, const float* ev,
                 int n, int slot) {
    const int E = n * 16;
    gemm_wmma<128, 64><<<(n + 127) / 128, 256, 0, stream>>>(X, W1, xw1, n);
    zero_f32<<<(n * 64 + 255) / 256, 256, 0, stream>>>(s1h, n * 64);
    long long tot1 = (long long)E * 64;
    scatter_add<<<(unsigned)((tot1 + 255) / 256), 256, 0, stream>>>(er, ec, ev, xw1, s1h, tot1, 6);
    bias_relu64<<<(n * 64 + 255) / 256, 256, 0, stream>>>(s1h, b1, n * 64);
    gemm_wmma<64, 16><<<(n + 127) / 128, 256, 0, stream>>>(s1h, W2, hw2, n);
    zero_f32<<<(n * 16 + 255) / 256, 256, 0, stream>>>(s2, n * 16);
    long long tot2 = (long long)E * 16;
    scatter_add<<<(unsigned)((tot2 + 255) / 256), 256, 0, stream>>>(er, ec, ev, hw2, s2, tot2, 4);
    pool_kernel<<<4, 256, 0, stream>>>(s2, b2, pools + slot * 64, n / 4);
  };

  run(X_su_ace, su_row,  su_col,  su_val,  N_SU_,  0);
  run(X_su_nmf, su_row,  su_col,  su_val,  N_SU_,  1);
  run(X_su_wat, su_row,  su_col,  su_val,  N_SU_,  2);
  run(X_sv_ace, ace_row, ace_col, ace_val, N_ACE_, 3);
  run(X_sv_nmf, nmf_row, nmf_col, nmf_val, N_NMF_, 4);
  run(X_sv_wat, wat_row, wat_col, wat_val, N_WAT_, 5);

  heads_kernel<<<1, 32, 0, stream>>>(pools, solute_env, ace_env, nmf_env, wat_env,
                                     fc1w, fc1b, fc2w, fc2b, fc3w, fc3b, fc4w, fc4b,
                                     (float*)d_out);
}
